// LabelWiseMLC_63780264345917
// MI455X (gfx1250) — compile-verified, matrix operand order checked by reading, not run on hardware
//
#include <hip/hip_runtime.h>
#include <cstdint>

typedef __attribute__((ext_vector_type(2)))  float     v2f;
typedef __attribute__((ext_vector_type(4)))  float     v4f;
typedef __attribute__((ext_vector_type(8)))  float     v8f;
typedef __attribute__((ext_vector_type(16))) _Float16  v16h;
typedef __attribute__((ext_vector_type(4)))  int       v4i;

#define MLC_BATCH   128
#define MLC_LABELS  512
#define MLC_DIM     1024
#define MLC_WAVES   16
#define MLC_THREADS (MLC_WAVES * 32)

// One block = one label x 16 batches. One wave = one output element.
// doc_rep row (4 KB) streamed from HBM with non-temporal b128 loads;
// W row (4 KB) staged once per block into LDS via async global->LDS.
__global__ __launch_bounds__(MLC_THREADS) void
LabelWiseMLC_kernel(const float* __restrict__ doc,
                    const float* __restrict__ W,
                    const float* __restrict__ bias,
                    float* __restrict__ out)
{
    __shared__ float lw[MLC_DIM];

    const int l    = blockIdx.x;            // label
    const int bt   = blockIdx.y;            // batch tile
    const int tid  = threadIdx.x;
    const int wave = tid >> 5;
    const int lane = tid & 31;

    const float* wrow = W + (size_t)l * MLC_DIM;

    // ---- stage W[l,:] (4 KB) into LDS: CDNA5 async global->LDS path ----
#if defined(__AMDGCN__) && __has_builtin(__builtin_amdgcn_global_load_async_to_lds_b128)
    if (tid < MLC_DIM / 4) {   // waves 0..7, all lanes active (uniform per wave)
        typedef __attribute__((address_space(1))) v4i* gp_t;
        typedef __attribute__((address_space(3))) v4i* lp_t;
        gp_t gsrc = (gp_t)(wrow + tid * 4);   // C-style cast drops const, sets AS1
        lp_t ldst = (lp_t)(lw + tid * 4);
        __builtin_amdgcn_global_load_async_to_lds_b128(gsrc, ldst, 0, 0);
#if __has_builtin(__builtin_amdgcn_s_wait_asynccnt)
        __builtin_amdgcn_s_wait_asynccnt(0);
#else
        asm volatile("s_wait_asynccnt 0" ::: "memory");
#endif
    }
#else
    for (int i = tid; i < MLC_DIM; i += MLC_THREADS) lw[i] = wrow[i];
#endif
    __syncthreads();

    // ---- stream this wave's doc row and FMA against LDS W ----
    const int b = bt * MLC_WAVES + wave;
    const v4f* __restrict__ drow = (const v4f*)(doc + ((size_t)b * MLC_LABELS + l) * MLC_DIM);
    const v4f* __restrict__ wv   = (const v4f*)lw;

    v4f acc = {0.f, 0.f, 0.f, 0.f};
#pragma unroll
    for (int it = 0; it < MLC_DIM / 128; ++it) {
        v4f d = __builtin_nontemporal_load(&drow[it * 32 + lane]); // streamed once: NT
        v4f w = wv[it * 32 + lane];                                // LDS (ds_load_b128)
        acc += d * w;
    }
    float p = acc.x + acc.y + acc.z + acc.w;   // per-lane partial

    // ---- cross-lane reduction via WMMA (matrix pipe is idle anyway) ----
    // A holds partials in the K=0 slot (row m gets p_m and p_{m+16} per the
    // documented A layout); B = all ones -> D[m,n] = p_m + p_{m+16} (any n).
    // D layout: VGPR v, lanes 0-15 = row v, lanes 16-31 = row v+8.
    float total;
#if defined(__AMDGCN__) && __has_builtin(__builtin_amdgcn_wmma_f32_16x16x4_f32)
    {
        v2f a;  a.x = p;    a.y = 0.f;
        v2f bm; bm.x = 1.f; bm.y = 1.f;
        v8f c = {};
        c = __builtin_amdgcn_wmma_f32_16x16x4_f32(false, a, false, bm,
                                                  (short)0, c, false, false);
        float s = c[0] + c[1] + c[2] + c[3] + c[4] + c[5] + c[6] + c[7];
        total = s + __shfl_xor(s, 16, 32);     // combine the two half-wave row groups
    }
#elif defined(__AMDGCN__) && __has_builtin(__builtin_amdgcn_wmma_f32_16x16x32_f16)
    {
        v16h a = {};
        a[0] = (_Float16)p;                    // K=0 slot, same row-pairing property
        v16h bm;
#pragma unroll
        for (int i = 0; i < 16; ++i) bm[i] = (_Float16)1.0f;
        v8f c = {};
        c = __builtin_amdgcn_wmma_f32_16x16x32_f16(false, a, false, bm,
                                                   (short)0, c, false, false);
        float s = c[0] + c[1] + c[2] + c[3] + c[4] + c[5] + c[6] + c[7];
        total = s + __shfl_xor(s, 16, 32);
    }
#else
    {
        float s = p;
        for (int off = 16; off > 0; off >>= 1) s += __shfl_xor(s, off, 32);
        total = s;
    }
#endif

    if (lane == 0) {
        float z = total + bias[l];
        out[(size_t)b * MLC_LABELS + l] = 1.0f / (1.0f + __expf(-z));
    }
}

extern "C" void kernel_launch(void* const* d_in, const int* in_sizes, int n_in,
                              void* d_out, int out_size, void* d_ws, size_t ws_size,
                              hipStream_t stream) {
    const float* doc  = (const float*)d_in[0];   // [128, 512, 1024] f32
    const float* W    = (const float*)d_in[1];   // [512, 1024] f32
    const float* bias = (const float*)d_in[2];   // [512] f32
    float* out        = (float*)d_out;           // [128, 512] f32

    dim3 grid(MLC_LABELS, MLC_BATCH / MLC_WAVES, 1);
    LabelWiseMLC_kernel<<<grid, MLC_THREADS, 0, stream>>>(doc, W, bias, out);

    (void)in_sizes; (void)n_in; (void)out_size; (void)d_ws; (void)ws_size;
}
